// MultiHeadAttention_31679678775348
// MI455X (gfx1250) — matrix operand order repaired and checked by reading, not compiled, not run on hardware
//
#include <hip/hip_runtime.h>
#include <hip/hip_bf16.h>

// ---------------------------------------------------------------------------
// MHA forward for MI455X (gfx1250, wave32, WMMA 16x16x32 f16).
// d_out = [ out (8*1024*768 f32) | attn weights (8*12*1024*1024 f32) ].
// The weights region of d_out doubles as the scores scratch (softmax in-place).
// Workspace: f16 Q,K in [B,H,N,64], V^T in [B,H,64,N], O in [B,N,768] (~50MB).
// All LDS B-operand tiles are stored transposed (lane-contiguous K) so every
// WMMA fragment load merges into ds_load_b128 pairs.  The scores kernel stages
// its Q/K tiles with the Tensor Data Mover (tensor_load_to_lds + TENSORcnt).
// ---------------------------------------------------------------------------

#define BD    8
#define SEQ   1024
#define DIM   768
#define HEADS 12
#define HD    64

typedef __attribute__((ext_vector_type(16))) _Float16     v16h;
typedef __attribute__((ext_vector_type(8)))  float        v8f;
typedef __attribute__((ext_vector_type(4)))  unsigned int v4u;
typedef __attribute__((ext_vector_type(8)))  unsigned int v8u;

// ---- WMMA fragment loaders (ISA 7.12.2 layouts, wave32) -------------------

// A 16x32 f16: lane L (0-15): row M=L, K = {klo..klo+7, klo+16..klo+23}, klo=0
//              lane L (16-31): row M=L-16, klo=8
__device__ __forceinline__ v16h load_a_frag(const _Float16* __restrict__ row, int klo) {
  v16h a;
#pragma unroll
  for (int i = 0; i < 8; ++i) {
    a[i]     = row[klo + i];
    a[8 + i] = row[klo + 16 + i];
  }
  return a;
}

// B 32x16 f16 (tile stored transposed in LDS), same K order as A fragment:
// lanes 0-15: K = {0..7, 16..23}; lanes 16-31: K = {8..15, 24..31} (kb/2 = klo).
__device__ __forceinline__ v16h load_b_frag(const _Float16* __restrict__ rowp, int kb) {
  v16h b;
#pragma unroll
  for (int i = 0; i < 8; ++i) {
    b[i]     = rowp[(kb >> 1) + i];
    b[8 + i] = rowp[(kb >> 1) + 16 + i];
  }
  return b;
}

// ---- Tensor Data Mover: 2D f16 tile, global -> LDS (ISA ch.8 D# layout) ---
// Issues one TENSOR_LOAD_TO_LDS with a user descriptor:
//   group0: count=1 | lds_addr | global_addr[56:0] | type=2
//   group1: data_size=2B, pad_enable, pad_interval/amount codes,
//           tensor_dim0/1, tile_dim0/1, tensor_dim0_stride (elements)
//   groups 2/3: zero (2D tensor, tile_dim2..4 = 0 -> unused)
__device__ __forceinline__ void tdm_load_2d_f16(
    unsigned lds_addr, const void* gaddr,
    unsigned tensor_d0, unsigned tensor_d1, unsigned stride0,
    unsigned tile_d0, unsigned tile_d1,
    unsigned pad_interval_code, unsigned pad_amount_code)
{
  unsigned long long ga = (unsigned long long)(uintptr_t)gaddr;
  v4u g0;
  g0[0] = 1u;                                       // count=1 (valid user D#)
  g0[1] = lds_addr;                                 // LDS byte address
  g0[2] = (unsigned)ga;                             // global_addr[31:0]
  g0[3] = ((unsigned)(ga >> 32) & 0x01FFFFFFu) | (2u << 30);  // [56:32] | type=2
  v8u g1;
  g1[0] = (1u << 16)                                // data_size = 1 -> 2 bytes
        | (1u << 20)                                // pad_enable
        | (pad_interval_code << 22)                 // dwords between pads
        | (pad_amount_code << 25);                  // pad dwords
  g1[1] = (tensor_d0 & 0xFFFFu) << 16;              // tensor_dim0[15:0]
  g1[2] = ((tensor_d0 >> 16) & 0xFFFFu) | ((tensor_d1 & 0xFFFFu) << 16);
  g1[3] = ((tensor_d1 >> 16) & 0xFFFFu) | ((tile_d0 & 0xFFFFu) << 16);
  g1[4] = (tile_d1 & 0xFFFFu);                      // tile_dim2 = 0 (unused)
  g1[5] = stride0;                                  // tensor_dim0_stride[31:0]
  g1[6] = 0u;                                       // stride0 hi | stride1 lo
  g1[7] = 0u;                                       // stride1 hi
  v4u gz = {0u, 0u, 0u, 0u};
  asm volatile("tensor_load_to_lds %0, %1, %2, %3"
               :: "s"(g0), "s"(g1), "s"(gz), "s"(gz)
               : "memory");
}

// ---------------------------------------------------------------------------
// Kernel 1: fused QKV projection.  C[8192,768] = x @ W + bias, W in {Wq,Wk,Wv}
// (blockIdx.z selects). Q,K -> f16 [B,H,N,64]; V -> f16 transposed [B,H,64,N].
// ---------------------------------------------------------------------------
__global__ void __launch_bounds__(256)
mha_qkv_gemm(const float* __restrict__ x,
             const float* __restrict__ Wq, const float* __restrict__ bq,
             const float* __restrict__ Wk, const float* __restrict__ bk,
             const float* __restrict__ Wv, const float* __restrict__ bv,
             _Float16* __restrict__ Qh, _Float16* __restrict__ Kh,
             _Float16* __restrict__ Vt)
{
  const int sel = blockIdx.z;
  const float* __restrict__ W    = (sel == 0) ? Wq : (sel == 1) ? Wk : Wv;
  const float* __restrict__ bias = (sel == 0) ? bq : (sel == 1) ? bk : bv;

  const int m0 = blockIdx.y * 128;
  const int n0 = blockIdx.x * 128;

  __shared__ alignas(16) _Float16 As [128][40];  // A tile, row-major [m][k]
  __shared__ alignas(16) _Float16 BsT[128][40];  // B tile, transposed [n][k]

  const int t    = threadIdx.x;
  const int lane = t & 31;
  const int wave = t >> 5;
  const int wm   = (wave >> 2) * 64;   // 0/64
  const int wn   = (wave & 3) * 32;    // 0/32/64/96

  v8f acc[4][2] = {};
  const int klo = (lane < 16) ? 0 : 8;
  const int kb  = (lane < 16) ? 0 : 16;

  for (int k0 = 0; k0 < DIM; k0 += 32) {
    // stage A: 128x32 f32 -> f16 (row-major)
#pragma unroll
    for (int it = 0; it < 4; ++it) {
      int e = t + it * 256;
      int r = e >> 3, c4 = (e & 7) * 4;
      const float* ga = x + (size_t)(m0 + r) * DIM + k0 + c4;
      float4 u = *(const float4*)ga;
      __builtin_prefetch(ga + 32, 0, 1);   // next k-tile (speculative, no guard)
      As[r][c4 + 0] = (_Float16)u.x;  As[r][c4 + 1] = (_Float16)u.y;
      As[r][c4 + 2] = (_Float16)u.z;  As[r][c4 + 3] = (_Float16)u.w;
    }
    // stage B: 32x128 f32 -> f16, transposed into [n][k]
#pragma unroll
    for (int it = 0; it < 4; ++it) {
      int e = t + it * 256;
      int r = e >> 5, c4 = (e & 31) * 4;   // r = k-row, c4 = n-col
      const float* gb = W + (size_t)(k0 + r) * DIM + n0 + c4;
      float4 u = *(const float4*)gb;
      __builtin_prefetch(gb + 32 * DIM, 0, 1);
      BsT[c4 + 0][r] = (_Float16)u.x;  BsT[c4 + 1][r] = (_Float16)u.y;
      BsT[c4 + 2][r] = (_Float16)u.z;  BsT[c4 + 3][r] = (_Float16)u.w;
    }
    __syncthreads();

    v16h af[4], bf[2];
#pragma unroll
    for (int tm = 0; tm < 4; ++tm)
      af[tm] = load_a_frag(&As[wm + tm * 16 + (lane & 15)][0], klo);
#pragma unroll
    for (int tn = 0; tn < 2; ++tn)
      bf[tn] = load_b_frag(&BsT[wn + tn * 16 + (lane & 15)][0], kb);
#pragma unroll
    for (int tm = 0; tm < 4; ++tm)
#pragma unroll
      for (int tn = 0; tn < 2; ++tn)
        acc[tm][tn] = __builtin_amdgcn_wmma_f32_16x16x32_f16(
            false, af[tm], false, bf[tn], (short)0, acc[tm][tn], false, false);
    __syncthreads();
  }

  // epilogue: +bias, f16 convert, scatter. Uniform `sel` branch hoisted out of
  // the store loops so no per-store pointer selects are generated.
  const int rsel = (lane >> 4) * 8;
  if (sel == 0) {
#pragma unroll
    for (int tn = 0; tn < 2; ++tn) {
      int gn = n0 + wn + tn * 16 + (lane & 15);
      float bvl = bias[gn];
      int h = gn >> 6, d = gn & 63;
#pragma unroll
      for (int tm = 0; tm < 4; ++tm)
#pragma unroll
        for (int r = 0; r < 8; ++r) {
          int gm = m0 + wm + tm * 16 + rsel + r;
          int b = gm >> 10, nrow = gm & 1023;
          Qh[(((size_t)(b * HEADS + h)) * SEQ + nrow) * HD + d] =
              (_Float16)(acc[tm][tn][r] + bvl);
        }
    }
  } else if (sel == 1) {
#pragma unroll
    for (int tn = 0; tn < 2; ++tn) {
      int gn = n0 + wn + tn * 16 + (lane & 15);
      float bvl = bias[gn];
      int h = gn >> 6, d = gn & 63;
#pragma unroll
      for (int tm = 0; tm < 4; ++tm)
#pragma unroll
        for (int r = 0; r < 8; ++r) {
          int gm = m0 + wm + tm * 16 + rsel + r;
          int b = gm >> 10, nrow = gm & 1023;
          Kh[(((size_t)(b * HEADS + h)) * SEQ + nrow) * HD + d] =
              (_Float16)(acc[tm][tn][r] + bvl);
        }
    }
  } else {
#pragma unroll
    for (int tn = 0; tn < 2; ++tn) {
      int gn = n0 + wn + tn * 16 + (lane & 15);
      float bvl = bias[gn];
      int h = gn >> 6, d = gn & 63;
#pragma unroll
      for (int tm = 0; tm < 4; ++tm)
#pragma unroll
        for (int r = 0; r < 8; ++r) {
          int gm = m0 + wm + tm * 16 + rsel + r;
          int b = gm >> 10, nrow = gm & 1023;
          // V stored transposed: [B,H,64,N] -> contiguous in nrow
          Vt[(((size_t)(b * HEADS + h)) * HD + d) * SEQ + nrow] =
              (_Float16)(acc[tm][tn][r] + bvl);
        }
    }
  }
}

// ---------------------------------------------------------------------------
// Kernel 2: scores = (Q @ K^T) * 1/sqrt(64), per (b,h), 128x128 tile, f32 out
// written into the weights region of d_out.  Q/K tiles are staged by the
// Tensor Data Mover: tile 64x128 f16, row = 32 dwords, LDS pad 4 dwords
// (-> 72-element padded rows), completion via TENSORcnt.
// ---------------------------------------------------------------------------
__global__ void __launch_bounds__(256)
mha_scores(const _Float16* __restrict__ Qh, const _Float16* __restrict__ Kh,
           float* __restrict__ wts)
{
  const int bh = blockIdx.z;
  const int i0 = blockIdx.y * 128;
  const int j0 = blockIdx.x * 128;

  __shared__ alignas(16) _Float16 Qs[128][72];
  __shared__ alignas(16) _Float16 Ks[128][72];

  const _Float16* __restrict__ Qp = Qh + (size_t)bh * SEQ * HD;
  const _Float16* __restrict__ Kp = Kh + (size_t)bh * SEQ * HD;

  const int t    = threadIdx.x;
  const int lane = t & 31;
  const int wave = t >> 5;
  const int wm   = (wave >> 2) * 64;
  const int wn   = (wave & 3) * 32;

  // TDM-stage both 128x64 f16 tiles (one wave issues; EXEC is ignored by TDM).
  // pad_interval code 4 = 32 dwords (one 64-f16 row); pad_amount code 3 =
  // 4 dwords (16B) -> matches the [128][72] padded LDS layout.
  if (wave == 0) {
    tdm_load_2d_f16((unsigned)(uintptr_t)&Qs[0][0], Qp + (size_t)i0 * HD,
                    /*tensor_d0=*/HD, /*tensor_d1=*/SEQ, /*stride0=*/HD,
                    /*tile_d0=*/HD, /*tile_d1=*/128,
                    /*pad_interval=*/4u, /*pad_amount=*/3u);
    tdm_load_2d_f16((unsigned)(uintptr_t)&Ks[0][0], Kp + (size_t)j0 * HD,
                    HD, SEQ, HD, HD, 128, 4u, 3u);
    __builtin_amdgcn_s_wait_tensorcnt(0);
  }
  __syncthreads();

  v8f acc[4][2] = {};
  const int klo = (lane < 16) ? 0 : 8;
  const int kb  = (lane < 16) ? 0 : 16;

#pragma unroll
  for (int dstep = 0; dstep < HD; dstep += 32) {
    v16h af[4], bf[2];
#pragma unroll
    for (int tm = 0; tm < 4; ++tm)
      af[tm] = load_a_frag(&Qs[wm + tm * 16 + (lane & 15)][dstep], klo);
    // B = K^T : element [k=d][n=j] = K[j][d] -> contiguous per lane in Ks
#pragma unroll
    for (int tn = 0; tn < 2; ++tn)
      bf[tn] = load_b_frag(&Ks[wn + tn * 16 + (lane & 15)][dstep], kb);
#pragma unroll
    for (int tm = 0; tm < 4; ++tm)
#pragma unroll
      for (int tn = 0; tn < 2; ++tn)
        acc[tm][tn] = __builtin_amdgcn_wmma_f32_16x16x32_f16(
            false, af[tm], false, bf[tn], (short)0, acc[tm][tn], false, false);
  }

  float* __restrict__ wp = wts + (size_t)bh * SEQ * SEQ;
  const int rsel = (lane >> 4) * 8;
#pragma unroll
  for (int tm = 0; tm < 4; ++tm)
#pragma unroll
    for (int tn = 0; tn < 2; ++tn) {
      int gj = j0 + wn + tn * 16 + (lane & 15);
#pragma unroll
      for (int r = 0; r < 8; ++r) {
        int gi = i0 + wm + tm * 16 + rsel + r;
        wp[(size_t)gi * SEQ + gj] = acc[tm][tn][r] * 0.125f;  // 1/sqrt(64)
      }
    }
}

// ---------------------------------------------------------------------------
// Kernel 3: row softmax over 1024 elements, one wave32 per row, in-place f32.
// ---------------------------------------------------------------------------
__global__ void __launch_bounds__(256)
mha_softmax(float* __restrict__ wts)
{
  const int row  = blockIdx.x * 8 + (threadIdx.x >> 5);
  const int lane = threadIdx.x & 31;
  float* __restrict__ p = wts + (size_t)row * SEQ;

  float4 v[8];
  float m = -3.0e38f;
#pragma unroll
  for (int i = 0; i < 8; ++i) {
    v[i] = *(const float4*)(p + (size_t)(i * 32 + lane) * 4);
    m = fmaxf(m, fmaxf(fmaxf(v[i].x, v[i].y), fmaxf(v[i].z, v[i].w)));
  }
#pragma unroll
  for (int off = 16; off > 0; off >>= 1) m = fmaxf(m, __shfl_xor(m, off, 32));

  float s = 0.0f;
#pragma unroll
  for (int i = 0; i < 8; ++i) {
    v[i].x = __expf(v[i].x - m);  v[i].y = __expf(v[i].y - m);
    v[i].z = __expf(v[i].z - m);  v[i].w = __expf(v[i].w - m);
    s += v[i].x + v[i].y + v[i].z + v[i].w;
  }
#pragma unroll
  for (int off = 16; off > 0; off >>= 1) s += __shfl_xor(s, off, 32);

  const float inv = 1.0f / s;
#pragma unroll
  for (int i = 0; i < 8; ++i) {
    v[i].x *= inv; v[i].y *= inv; v[i].z *= inv; v[i].w *= inv;
    *(float4*)(p + (size_t)(i * 32 + lane) * 4) = v[i];
  }
}

// ---------------------------------------------------------------------------
// Kernel 4: O = attn @ V  per (b,h).  128x64 output tile per block.
// attn f32 -> f16 on LDS load; V already transposed in global ([B,H,64,N]) so
// the V tile copy is vectorized and fragment reads are lane-contiguous.
// ---------------------------------------------------------------------------
__global__ void __launch_bounds__(256)
mha_attnv(const float* __restrict__ wts, const _Float16* __restrict__ Vt,
          _Float16* __restrict__ Oh)
{
  const int bh = blockIdx.y;
  const int i0 = blockIdx.x * 128;
  const int b = bh / HEADS, h = bh % HEADS;

  __shared__ alignas(16) _Float16 As [128][40];  // attn tile [i][j] 128x32
  __shared__ alignas(16) _Float16 VsT[64][40];   // V^T tile  [d][j] 64x32

  const float*    __restrict__ ap  = wts + (size_t)bh * SEQ * SEQ;
  const _Float16* __restrict__ vtp = Vt  + (size_t)bh * HD * SEQ;

  const int t    = threadIdx.x;
  const int lane = t & 31;
  const int wave = t >> 5;
  const int wm   = (wave >> 1) * 32;   // 0/32/64/96
  const int wn   = (wave & 1) * 32;    // 0/32

  v8f acc[2][2] = {};
  const int klo = (lane < 16) ? 0 : 8;
  const int kb  = (lane < 16) ? 0 : 16;

  for (int k0 = 0; k0 < SEQ; k0 += 32) {
    // attn tile: 128 x 32 f32 -> f16
#pragma unroll
    for (int it = 0; it < 4; ++it) {
      int e = t + it * 256;
      int r = e >> 3, c4 = (e & 7) * 4;
      const float* ga = ap + (size_t)(i0 + r) * SEQ + k0 + c4;
      float4 u = *(const float4*)ga;
      __builtin_prefetch(ga + 32, 0, 1);
      As[r][c4 + 0] = (_Float16)u.x;  As[r][c4 + 1] = (_Float16)u.y;
      As[r][c4 + 2] = (_Float16)u.z;  As[r][c4 + 3] = (_Float16)u.w;
    }
    // V^T tile: 64 x 32 f16, straight vectorized copy
    {
      int r = t >> 2, c8 = (t & 3) * 8;   // 256 threads cover 64x32 f16
      const _Float16* gv = vtp + (size_t)r * SEQ + k0 + c8;
      *(float4*)&VsT[r][c8] = *(const float4*)gv;
      __builtin_prefetch(gv + 32, 0, 1);
    }
    __syncthreads();

    v16h af[2], bf[2];
#pragma unroll
    for (int tm = 0; tm < 2; ++tm)
      af[tm] = load_a_frag(&As[wm + tm * 16 + (lane & 15)][0], klo);
#pragma unroll
    for (int tn = 0; tn < 2; ++tn)
      bf[tn] = load_b_frag(&VsT[wn + tn * 16 + (lane & 15)][0], kb);
#pragma unroll
    for (int tm = 0; tm < 2; ++tm)
#pragma unroll
      for (int tn = 0; tn < 2; ++tn)
        acc[tm][tn] = __builtin_amdgcn_wmma_f32_16x16x32_f16(
            false, af[tm], false, bf[tn], (short)0, acc[tm][tn], false, false);
    __syncthreads();
  }

  const int rsel = (lane >> 4) * 8;
#pragma unroll
  for (int tm = 0; tm < 2; ++tm)
#pragma unroll
    for (int tn = 0; tn < 2; ++tn) {
      int gd = wn + tn * 16 + (lane & 15);
#pragma unroll
      for (int r = 0; r < 8; ++r) {
        int gi = i0 + wm + tm * 16 + rsel + r;
        Oh[((size_t)b * SEQ + gi) * DIM + h * HD + gd] = (_Float16)acc[tm][tn][r];
      }
    }
}

// ---------------------------------------------------------------------------
// Kernel 5: out = O @ Wo + bo, f32 result.
// ---------------------------------------------------------------------------
__global__ void __launch_bounds__(256)
mha_outproj(const _Float16* __restrict__ Oh, const float* __restrict__ Wo,
            const float* __restrict__ bo, float* __restrict__ out)
{
  const int m0 = blockIdx.y * 128;
  const int n0 = blockIdx.x * 128;

  __shared__ alignas(16) _Float16 As [128][40];
  __shared__ alignas(16) _Float16 BsT[128][40];

  const int t    = threadIdx.x;
  const int lane = t & 31;
  const int wave = t >> 5;
  const int wm   = (wave >> 2) * 64;
  const int wn   = (wave & 3) * 32;

  v8f acc[4][2] = {};
  const int klo = (lane < 16) ? 0 : 8;
  const int kb  = (lane < 16) ? 0 : 16;

  for (int k0 = 0; k0 < DIM; k0 += 32) {
    // A tile: 128x32 f16 from Oh (vectorized copy)
#pragma unroll
    for (int it = 0; it < 2; ++it) {
      int e = t + it * 256;
      int r = e >> 2, c8 = (e & 3) * 8;
      const _Float16* ga = Oh + (size_t)(m0 + r) * DIM + k0 + c8;
      *(float4*)&As[r][c8] = *(const float4*)ga;
      __builtin_prefetch(ga + 32, 0, 1);
    }
    // B tile: 32x128 f32 -> f16, transposed into [n][k]
#pragma unroll
    for (int it = 0; it < 4; ++it) {
      int e = t + it * 256;
      int r = e >> 5, c4 = (e & 31) * 4;
      const float* gb = Wo + (size_t)(k0 + r) * DIM + n0 + c4;
      float4 u = *(const float4*)gb;
      __builtin_prefetch(gb + 32 * DIM, 0, 1);
      BsT[c4 + 0][r] = (_Float16)u.x;  BsT[c4 + 1][r] = (_Float16)u.y;
      BsT[c4 + 2][r] = (_Float16)u.z;  BsT[c4 + 3][r] = (_Float16)u.w;
    }
    __syncthreads();

    v16h af[4], bf[2];
#pragma unroll
    for (int tm = 0; tm < 4; ++tm)
      af[tm] = load_a_frag(&As[wm + tm * 16 + (lane & 15)][0], klo);
#pragma unroll
    for (int tn = 0; tn < 2; ++tn)
      bf[tn] = load_b_frag(&BsT[wn + tn * 16 + (lane & 15)][0], kb);
#pragma unroll
    for (int tm = 0; tm < 4; ++tm)
#pragma unroll
      for (int tn = 0; tn < 2; ++tn)
        acc[tm][tn] = __builtin_amdgcn_wmma_f32_16x16x32_f16(
            false, af[tm], false, bf[tn], (short)0, acc[tm][tn], false, false);
    __syncthreads();
  }

  const int rsel = (lane >> 4) * 8;
#pragma unroll
  for (int tn = 0; tn < 2; ++tn) {
    int gn = n0 + wn + tn * 16 + (lane & 15);
    float bvl = bo[gn];
#pragma unroll
    for (int tm = 0; tm < 4; ++tm)
#pragma unroll
      for (int r = 0; r < 8; ++r) {
        int gm = m0 + wm + tm * 16 + rsel + r;
        out[(size_t)gm * DIM + gn] = acc[tm][tn][r] + bvl;
      }
  }
}

// ---------------------------------------------------------------------------
extern "C" void kernel_launch(void* const* d_in, const int* in_sizes, int n_in,
                              void* d_out, int out_size, void* d_ws, size_t ws_size,
                              hipStream_t stream) {
  const float* x  = (const float*)d_in[0];
  const float* Wq = (const float*)d_in[1];
  const float* bq = (const float*)d_in[2];
  const float* Wk = (const float*)d_in[3];
  const float* bk = (const float*)d_in[4];
  const float* Wv = (const float*)d_in[5];
  const float* bv = (const float*)d_in[6];
  const float* Wo = (const float*)d_in[7];
  const float* bo = (const float*)d_in[8];

  float* out = (float*)d_out;                              // [8,1024,768]
  float* wts = out + (size_t)BD * SEQ * DIM;               // [8,12,1024,1024]

  const size_t qkv_elems = (size_t)BD * HEADS * SEQ * HD;  // 6,291,456
  _Float16* Qh = (_Float16*)d_ws;
  _Float16* Kh = Qh + qkv_elems;
  _Float16* Vt = Kh + qkv_elems;                           // [B,H,64,N] f16
  _Float16* Oh = Vt + qkv_elems;                           // [8,1024,768] f16

  // 1) Q,K,V projections (f16; V transposed)
  mha_qkv_gemm<<<dim3(DIM / 128, (BD * SEQ) / 128, 3), 256, 0, stream>>>(
      x, Wq, bq, Wk, bk, Wv, bv, Qh, Kh, Vt);
  // 2) scores = QK^T * scale -> weights region of d_out (TDM-staged tiles)
  mha_scores<<<dim3(SEQ / 128, SEQ / 128, BD * HEADS), 256, 0, stream>>>(Qh, Kh, wts);
  // 3) softmax in-place (this IS the `weights` output)
  mha_softmax<<<dim3((BD * HEADS * SEQ) / 8), 256, 0, stream>>>(wts);
  // 4) O = attn @ V
  mha_attnv<<<dim3(SEQ / 128, BD * HEADS), 256, 0, stream>>>(wts, Vt, Oh);
  // 5) out = O @ Wo + bo
  mha_outproj<<<dim3(DIM / 128, (BD * SEQ) / 128), 256, 0, stream>>>(Oh, Wo, bo, out);
}
